// LGGAT_68015102099730
// MI455X (gfx1250) — matrix operand order repaired and checked by reading, not compiled, not run on hardware
//
#include <hip/hip_runtime.h>

// ---------------------------------------------------------------------------
// LGGAT forward for MI455X (gfx1250, wave32, WMMA).
// Big GEMMs run in bf16 WMMA (v_wmma_f32_16x16x32_bf16) with f32 accumulate;
// softmax / scatter / update stay fp32. a_b2 is dropped exactly (softmax is
// shift-invariant). LDS staging uses GLOBAL_LOAD_ASYNC_TO_LDS_B128 when the
// toolchain exposes it (ASYNCcnt-tracked), else 16B vector copies.
// ---------------------------------------------------------------------------

typedef float  v8f   __attribute__((ext_vector_type(8)));
typedef __bf16 v16bf __attribute__((ext_vector_type(16)));
typedef __bf16 v8bf  __attribute__((ext_vector_type(8)));
typedef int    v4i   __attribute__((vector_size(16)));  // matches builtin sig

typedef __attribute__((address_space(1))) v4i* gbl_v4i_ptr;
typedef __attribute__((address_space(3))) v4i* lds_v4i_ptr;

#if __has_builtin(__builtin_amdgcn_global_load_async_to_lds_b128) && \
    __has_builtin(__builtin_amdgcn_s_wait_asynccnt)
#define HAVE_ASYNC_LDS 1
#else
#define HAVE_ASYNC_LDS 0
#endif

// Copy 16 bytes global -> LDS. Async path writes LDS directly via the
// async pipe (ASYNCcnt-tracked); fallback is a b128 load + ds_store_b128.
__device__ __forceinline__ void stage16(void* lds_dst, const void* gsrc) {
#if HAVE_ASYNC_LDS
  __builtin_amdgcn_global_load_async_to_lds_b128(
      (gbl_v4i_ptr)gsrc, (lds_v4i_ptr)lds_dst, 0, 0);
#else
  *(uint4*)lds_dst = *(const uint4*)gsrc;
#endif
}

__device__ __forceinline__ void stage_wait() {
#if HAVE_ASYNC_LDS
  __builtin_amdgcn_s_wait_asynccnt(0);
#endif
}

__device__ __forceinline__ v8f wmma_bf16(v16bf a, v16bf b, v8f c) {
  // 8 args: (neg_a, A, neg_b, B, c_mod, C, reuse_a, reuse_b)
  return __builtin_amdgcn_wmma_f32_16x16x32_bf16(false, a, false, b, (short)0,
                                                 c, false, false);
}

__device__ __forceinline__ v16bf cat16(v8bf lo, v8bf hi) {
  return __builtin_shufflevector(lo, hi, 0, 1, 2, 3, 4, 5, 6, 7, 8, 9, 10, 11,
                                 12, 13, 14, 15);
}

// Order-preserving float<->uint mapping so we can use atomicMax(uint*).
__device__ __forceinline__ unsigned f2ord(float f) {
  unsigned b = __float_as_uint(f);
  return (b & 0x80000000u) ? ~b : (b | 0x80000000u);
}
__device__ __forceinline__ float ord2f(unsigned u) {
  unsigned b = (u & 0x80000000u) ? (u ^ 0x80000000u) : ~u;
  return __uint_as_float(b);
}

// ---------------------------------------------------------------------------
// Elementwise converts / transposes (run once per launch; tiny vs GEMMs)
// ---------------------------------------------------------------------------
__global__ void cvt_bf16_kernel(const float* __restrict__ in,
                                __bf16* __restrict__ out, long n) {
  long i = (long)blockIdx.x * blockDim.x + threadIdx.x;
  if (i < n) out[i] = (__bf16)in[i];
}

// in: batch x [K][Nc] row-major f32 ; out: batch x [Nc][K] bf16 (col-major)
__global__ void transpose_cvt_kernel(const float* __restrict__ in,
                                     __bf16* __restrict__ out, int K, int Nc,
                                     long total) {
  long i = (long)blockIdx.x * blockDim.x + threadIdx.x;
  if (i >= total) return;
  long mat = (long)K * Nc;
  long b = i / mat;
  long r = i - b * mat;
  int k = (int)(r / Nc);
  int n = (int)(r - (long)k * Nc);
  out[b * mat + (long)n * K + k] = (__bf16)in[i];
}

// ---------------------------------------------------------------------------
// Per-layer scratch init
// ---------------------------------------------------------------------------
__global__ void init_kernel(float* __restrict__ a_arr, int E,
                            unsigned* __restrict__ amax,
                            float* __restrict__ denom, float* __restrict__ hv,
                            int N, long NH) {
  long i = (long)blockIdx.x * blockDim.x + threadIdx.x;
  if (i < NH) hv[i] = 0.0f;
  if (i < E) a_arr[i] = 0.0f;
  if (i < N) {
    amax[i] = 0u;  // == encoding floor (below enc(-inf))
    denom[i] = 0.0f;
  }
}

// ---------------------------------------------------------------------------
// Dense GEMM + bias: out[N,H] = A[N,K](bf16) * B[K,H] + bias, B given
// col-major bf16 (BT[H][K]). One wave per 16-row tile.
// ---------------------------------------------------------------------------
__global__ __launch_bounds__(256) void gemm_bias_kernel(
    const __bf16* __restrict__ Abf, const __bf16* __restrict__ BT,
    const float* __restrict__ bias, float* __restrict__ out, int N, int K,
    int H) {
  int gwave = (int)((blockIdx.x * (long)blockDim.x + threadIdx.x) >> 5);
  int rtiles = (N + 15) >> 4;
  if (gwave >= rtiles) return;  // wave-uniform: EXEC all-ones inside
  int lane = threadIdx.x & 31;
  int m = lane & 15;
  int hf = lane >> 4;
  int row = gwave * 16 + m;
  int rA = row < N ? row : N - 1;
  const __bf16* arow = Abf + (size_t)rA * K;
  for (int nt = 0; nt < (H >> 4); ++nt) {
    int col = nt * 16 + m;
    const __bf16* bcol = BT + (size_t)col * K;
    v8f acc = {};
    for (int k0 = 0; k0 < K; k0 += 32) {
      v8bf alo = *(const v8bf*)&arow[k0 + 16 * hf];
      v8bf ahi = *(const v8bf*)&arow[k0 + 16 * hf + 8];
      v16bf b = *(const v16bf*)&bcol[k0 + 16 * hf];
      acc = wmma_bf16(cat16(alo, ahi), b, acc);
    }
    float bb = bias[col];
#pragma unroll
    for (int i = 0; i < 8; ++i) {
      int r = gwave * 16 + i + 8 * hf;  // C layout: vgpr i -> row i + 8*half
      if (r < N) out[(size_t)r * H + col] = acc[i] + bb;
    }
  }
}

// ---------------------------------------------------------------------------
// Fused per-edge attention logits:
//   a[e] += sum_n relu( cat(h[src],h[dst],gh)[e,:] @ W1[:,n] + b1[n] ) * w2[n]
// Workgroup: 256 threads = 8 waves, 64 edges.  LDS: cat tile (bf16) + W1
// K-chunk (bf16, col-major).  Assumes H==256 (fixed problem size).
// ---------------------------------------------------------------------------
#define AK_EDGES 64
#define AK_KC 128

__global__ __launch_bounds__(256) void attn_kernel(
    const __bf16* __restrict__ h_bf, const float* __restrict__ gh,
    const int* __restrict__ src, const int* __restrict__ dst,
    const __bf16* __restrict__ W1T, const float* __restrict__ b1,
    const float* __restrict__ w2, float* __restrict__ a_arr, int E, int H) {
  __shared__ alignas(32) __bf16 sA[AK_EDGES * AK_KC];  // 16 KB
  __shared__ alignas(32) __bf16 sB[256 * AK_KC];       // 64 KB
  const int tid = threadIdx.x;
  const int lane = tid & 31;
  const int wave = tid >> 5;
  const int m = lane & 15;
  const int hf = lane >> 4;
  const int mt = wave & 3;   // which 16-edge m-tile
  const int nh = wave >> 2;  // column half: [nh*128, nh*128+128)
  const int e_base = blockIdx.x * AK_EDGES;
  const int Ktot = 3 * H;

  v8f acc[8];
#pragma unroll
  for (int i = 0; i < 8; ++i) acc[i] = (v8f){};

  for (int kc = 0; kc < Ktot; kc += AK_KC) {
    // ---- stage cat tile: 64 edges x 128 k (bf16). Each 128-chunk lies
    // entirely in one segment of the concat (H==256, AK_KC==128).
    {
      int el = tid >> 2;   // edge-local 0..63
      int part = tid & 3;  // 4 threads x 32 elems (64B) each
      int e = e_base + el;
      if (e >= E) e = E - 1;  // clamp; results discarded by epilogue guard
      int seg = kc / H;
      int off = (kc - seg * H) + part * 32;
      __bf16* dp = &sA[el * AK_KC + part * 32];
      if (seg == 2) {
        // gh is f32 in HBM: convert while staging (can't use async DMA)
        const float* p = gh + (size_t)e * H + off;
#pragma unroll
        for (int j = 0; j < 32; ++j) dp[j] = (__bf16)p[j];
      } else {
        const __bf16* p =
            h_bf + (size_t)(seg == 0 ? src[e] : dst[e]) * H + off;
#pragma unroll
        for (int j = 0; j < 4; ++j) stage16(dp + j * 8, p + j * 8);
      }
    }
    // ---- stage W1 chunk col-major: 256 cols x 128 k (16B chunks)
    for (int t = tid; t < 256 * (AK_KC / 8); t += 256) {
      int n = t >> 4;  // 16 v8bf chunks per column
      int k8 = (t & 15) * 8;
      stage16(&sB[n * AK_KC + k8], &W1T[(size_t)n * Ktot + kc + k8]);
    }
    stage_wait();     // ASYNCcnt==0: our LDS writes have landed
    __syncthreads();  // ... and are visible to the whole workgroup
    // ---- 4 WMMA k-steps of 32, 8 n-tiles per wave
#pragma unroll
    for (int ks = 0; ks < AK_KC / 32; ++ks) {
      int k0 = ks * 32;
      const __bf16* ap = &sA[(mt * 16 + m) * AK_KC + k0];
      v16bf a = cat16(*(const v8bf*)&ap[16 * hf], *(const v8bf*)&ap[16 * hf + 8]);
#pragma unroll
      for (int nt = 0; nt < 8; ++nt) {
        int col = nh * 128 + nt * 16 + m;
        v16bf b = *(const v16bf*)&sB[col * AK_KC + k0 + 16 * hf];
        acc[nt] = wmma_bf16(a, b, acc[nt]);
      }
    }
    __syncthreads();
  }

  // ---- fused epilogue: bias + relu + dot(w2), then 16-lane butterfly.
  float rs[8] = {0, 0, 0, 0, 0, 0, 0, 0};
#pragma unroll
  for (int nt = 0; nt < 8; ++nt) {
    int col = nh * 128 + nt * 16 + m;
    float bb = b1[col], wv = w2[col];
#pragma unroll
    for (int i = 0; i < 8; ++i) {
      float v = acc[nt][i] + bb;
      v = v > 0.0f ? v : 0.0f;
      rs[i] += v * wv;
    }
  }
#pragma unroll
  for (int off = 1; off < 16; off <<= 1)
#pragma unroll
    for (int i = 0; i < 8; ++i) rs[i] += __shfl_xor(rs[i], off, 32);
  if (m == 0) {  // lanes 0 (rows 0..7) and 16 (rows 8..15)
#pragma unroll
    for (int i = 0; i < 8; ++i) {
      int e = e_base + mt * 16 + i + 8 * hf;
      if (e < E) atomicAdd(&a_arr[e], rs[i]);  // two waves per edge -> atomic
    }
  }
}

// ---------------------------------------------------------------------------
// edge_softmax pieces
// ---------------------------------------------------------------------------
__global__ void amax_kernel(const float* __restrict__ a_arr,
                            const int* __restrict__ dst,
                            unsigned* __restrict__ amax, int E) {
  int e = blockIdx.x * blockDim.x + threadIdx.x;
  if (e < E) atomicMax(&amax[dst[e]], f2ord(a_arr[e]));
}

__global__ void ea_kernel(const float* __restrict__ a_arr,
                          const int* __restrict__ dst,
                          const unsigned* __restrict__ amax,
                          float* __restrict__ ea_arr,
                          float* __restrict__ denom, int E) {
  int e = blockIdx.x * blockDim.x + threadIdx.x;
  if (e >= E) return;
  int d = dst[e];
  float v = expf(a_arr[e] - ord2f(amax[d]));
  ea_arr[e] = v;
  atomicAdd(&denom[d], v);
}

// message scatter: hv[dst] += att * h[src] * gh   (4 edges / block, H==256)
__global__ __launch_bounds__(256) void msg_kernel(
    const float* __restrict__ h, const float* __restrict__ gh,
    const int* __restrict__ src, const int* __restrict__ dst,
    const float* __restrict__ ea, const float* __restrict__ denom,
    float* __restrict__ hv, int E, int H) {
  int e = blockIdx.x * 4 + (threadIdx.x >> 6);
  if (e >= E) return;
  int c = (threadIdx.x & 63) * 4;
  int d = dst[e], s = src[e];
  float att = ea[e] / denom[d];
  float4 hs = *(const float4*)&h[(size_t)s * H + c];
  float4 g4 = *(const float4*)&gh[(size_t)e * H + c];
  float* hp = &hv[(size_t)d * H + c];
  atomicAdd(hp + 0, att * hs.x * g4.x);
  atomicAdd(hp + 1, att * hs.y * g4.y);
  atomicAdd(hp + 2, att * hs.z * g4.z);
  atomicAdd(hp + 3, att * hs.w * g4.w);
}

// ---------------------------------------------------------------------------
// eps = relu(h @ W1 + b1) @ w2 + b2 per node (WMMA, fused epilogue)
// ---------------------------------------------------------------------------
__global__ __launch_bounds__(256) void eps_kernel(
    const __bf16* __restrict__ h_bf, const __bf16* __restrict__ W1T,
    const float* __restrict__ b1, const float* __restrict__ w2,
    const float* __restrict__ b2, float* __restrict__ eps_out, int N, int H) {
  int gwave = (int)((blockIdx.x * (long)blockDim.x + threadIdx.x) >> 5);
  int rtiles = (N + 15) >> 4;
  if (gwave >= rtiles) return;
  int lane = threadIdx.x & 31;
  int m = lane & 15;
  int hf = lane >> 4;
  int row = gwave * 16 + m;
  int rA = row < N ? row : N - 1;
  const __bf16* arow = h_bf + (size_t)rA * H;
  float rs[8] = {0, 0, 0, 0, 0, 0, 0, 0};
  for (int nt = 0; nt < (H >> 4); ++nt) {
    int col = nt * 16 + m;
    const __bf16* bcol = W1T + (size_t)col * H;
    v8f acc = {};
    for (int k0 = 0; k0 < H; k0 += 32) {
      v8bf alo = *(const v8bf*)&arow[k0 + 16 * hf];
      v8bf ahi = *(const v8bf*)&arow[k0 + 16 * hf + 8];
      v16bf b = *(const v16bf*)&bcol[k0 + 16 * hf];
      acc = wmma_bf16(cat16(alo, ahi), b, acc);
    }
    float bb = b1[col], wv = w2[col];
#pragma unroll
    for (int i = 0; i < 8; ++i) {
      float v = acc[i] + bb;
      v = v > 0.0f ? v : 0.0f;
      rs[i] += v * wv;
    }
  }
#pragma unroll
  for (int off = 1; off < 16; off <<= 1)
#pragma unroll
    for (int i = 0; i < 8; ++i) rs[i] += __shfl_xor(rs[i], off, 32);
  if (m == 0) {
    float bb2 = b2[0];
#pragma unroll
    for (int i = 0; i < 8; ++i) {
      int r = gwave * 16 + i + 8 * hf;
      if (r < N) eps_out[r] = rs[i] + bb2;
    }
  }
}

// h_out = (1 + eps) * hv + h_in
__global__ void update_kernel(const float* __restrict__ h_in,
                              const float* __restrict__ hv,
                              const float* __restrict__ eps,
                              float* __restrict__ h_out, long NH, int H) {
  long i = (long)blockIdx.x * blockDim.x + threadIdx.x;
  if (i >= NH) return;
  long r = i / H;
  h_out[i] = (1.0f + eps[r]) * hv[i] + h_in[i];
}

// ---------------------------------------------------------------------------
// Host
// ---------------------------------------------------------------------------
static inline long cdiv(long a, long b) { return (a + b - 1) / b; }

extern "C" void kernel_launch(void* const* d_in, const int* in_sizes, int n_in,
                              void* d_out, int out_size, void* d_ws,
                              size_t ws_size, hipStream_t stream) {
  (void)n_in; (void)out_size; (void)ws_size;
  const float* node_feats = (const float*)d_in[0];
  const float* gh = (const float*)d_in[1];
  const int* src = (const int*)d_in[2];
  const int* dst = (const int*)d_in[3];
  const float* Wnw = (const float*)d_in[4];
  const float* Wnb = (const float*)d_in[5];
  const float* eW1 = (const float*)d_in[6];
  const float* eB1 = (const float*)d_in[7];
  const float* eW2 = (const float*)d_in[8];
  const float* eB2 = (const float*)d_in[9];
  const float* aW1 = (const float*)d_in[10];
  const float* aB1 = (const float*)d_in[11];
  const float* aW2 = (const float*)d_in[12];
  (void)d_in[13];  // a_b2: constant shift inside edge_softmax -> exactly no-op

  const int H = in_sizes[5];          // 256
  const int D = in_sizes[4] / H;      // 128
  const int N = in_sizes[0] / D;      // 10000
  const int E = in_sizes[2];          // 160000
  const int L = in_sizes[7] / H;      // 3
  const long NH = (long)N * H;
  const int K3 = 3 * H;

  // ---- workspace carve (all buffers we read are initialized every call)
  char* w = (char*)d_ws;
  auto carve = [&](size_t bytes) {
    char* p = w;
    w += (bytes + 255) & ~(size_t)255;
    return p;
  };
  float* a_arr = (float*)carve((size_t)E * 4);
  float* ea_arr = (float*)carve((size_t)E * 4);
  unsigned* amax = (unsigned*)carve((size_t)N * 4);
  float* denom = (float*)carve((size_t)N * 4);
  float* hv = (float*)carve((size_t)NH * 4);
  float* eps_arr = (float*)carve((size_t)N * 4);
  __bf16* h_bf = (__bf16*)carve((size_t)NH * 2);
  __bf16* node_bf = (__bf16*)carve((size_t)N * D * 2);
  __bf16* WnT = (__bf16*)carve((size_t)H * D * 2);
  __bf16* aW1T = (__bf16*)carve((size_t)L * K3 * H * 2);
  __bf16* eW1T = (__bf16*)carve((size_t)L * H * H * 2);

  // ---- one-time converts / transposes (bf16, col-major weights)
  cvt_bf16_kernel<<<cdiv((long)N * D, 256), 256, 0, stream>>>(node_feats,
                                                              node_bf,
                                                              (long)N * D);
  transpose_cvt_kernel<<<cdiv((long)D * H, 256), 256, 0, stream>>>(
      Wnw, WnT, D, H, (long)D * H);
  transpose_cvt_kernel<<<cdiv((long)L * K3 * H, 256), 256, 0, stream>>>(
      aW1, aW1T, K3, H, (long)L * K3 * H);
  transpose_cvt_kernel<<<cdiv((long)L * H * H, 256), 256, 0, stream>>>(
      eW1, eW1T, H, H, (long)L * H * H);

  const int rtiles = (N + 15) / 16;
  const int gemm_blocks = (int)cdiv(rtiles, 8);

  // ---- layer 0 output: input projection
  gemm_bias_kernel<<<gemm_blocks, 256, 0, stream>>>(node_bf, WnT, Wnb,
                                                    (float*)d_out, N, D, H);

  long init_n = NH > E ? NH : E;
  for (int l = 0; l < L; ++l) {
    const float* h_in = (const float*)d_out + (size_t)l * NH;
    float* h_out = (float*)d_out + (size_t)(l + 1) * NH;

    cvt_bf16_kernel<<<cdiv(NH, 256), 256, 0, stream>>>(h_in, h_bf, NH);
    init_kernel<<<cdiv(init_n, 256), 256, 0, stream>>>(a_arr, E, amax, denom,
                                                       hv, N, NH);
    attn_kernel<<<cdiv(E, AK_EDGES), 256, 0, stream>>>(
        h_bf, gh, src, dst, aW1T + (size_t)l * K3 * H, aB1 + (size_t)l * H,
        aW2 + (size_t)l * H, a_arr, E, H);
    amax_kernel<<<cdiv(E, 256), 256, 0, stream>>>(a_arr, dst, amax, E);
    ea_kernel<<<cdiv(E, 256), 256, 0, stream>>>(a_arr, dst, amax, ea_arr,
                                                denom, E);
    msg_kernel<<<cdiv(E, 4), 256, 0, stream>>>(h_in, gh, src, dst, ea_arr,
                                               denom, hv, E, H);
    eps_kernel<<<gemm_blocks, 256, 0, stream>>>(
        h_bf, eW1T + (size_t)l * H * H, eB1 + (size_t)l * H,
        eW2 + (size_t)l * H, eB2 + l, eps_arr, N, H);
    update_kernel<<<cdiv(NH, 256), 256, 0, stream>>>(h_in, hv, eps_arr, h_out,
                                                     NH, H);
  }
}
